// GCN_JK_Model_74337293959433
// MI455X (gfx1250) — matrix operand order repaired and checked by reading, not compiled, not run on hardware
//
#include <hip/hip_runtime.h>
#include <hip/hip_bf16.h>

typedef __attribute__((ext_vector_type(16))) __bf16 v16bf;
typedef __attribute__((ext_vector_type(8)))  float  v8f;

#define HDIM 128   // hidden size
#define XSW  384   // JK concat width (3*HDIM)

// ---------------- degree / norm kernels ----------------

__global__ void deg_init_kernel(float* __restrict__ deg, int n) {
  int i = blockIdx.x * blockDim.x + threadIdx.x;
  if (i < n) deg[i] = 1.0f;  // self-loop contribution
}

__global__ void deg_count_kernel(const int* __restrict__ col, float* __restrict__ deg, int e) {
  int i = blockIdx.x * blockDim.x + threadIdx.x;
  if (i < e) unsafeAtomicAdd(&deg[col[i]], 1.0f);
}

__global__ void deg_rsqrt_kernel(float* __restrict__ deg, int n) {
  int i = blockIdx.x * blockDim.x + threadIdx.x;
  if (i < n) deg[i] = rsqrtf(deg[i]);  // deg >= 1 always (self-loops)
}

// ---------------- weight prep: W (K x Hc fp32) -> Wt (Hc x K bf16) ----------------
__global__ void wconv_kernel(const float* __restrict__ W, __bf16* __restrict__ Wt,
                             int K, int Hc) {
  int idx = blockIdx.x * blockDim.x + threadIdx.x;
  if (idx >= K * Hc) return;
  int k = idx / Hc, n = idx - k * Hc;
  Wt[n * K + k] = (__bf16)W[idx];
}

// ---------------- WMMA GEMM: C[16-row strip] = A @ Wt^T (+bias) ----------------
// A: fp32 row-major (32-bit indexing). Wt: Hc x K bf16 row-major (pre-transposed).
// A tile staged memory->LDS with GLOBAL_LOAD_ASYNC_TO_LDS_B128 (ASYNCcnt path).
template<int NWAVES, bool ADD_BIAS>
__global__ __launch_bounds__(NWAVES * 32)
void wmma_gemm_kernel(const float* __restrict__ A, int lda,
                      const __bf16* __restrict__ Wt,
                      const float* __restrict__ bias,
                      float* __restrict__ C, int ldc,
                      int K, int Hc)
{
  __shared__ float As[16][36];           // stride 144B: 16B-aligned b128 dests
  constexpr int NT = NWAVES * 32;
  const int tid  = threadIdx.x;
  const int lane = tid & 31;
  const int wave = tid >> 5;
  const int half = lane >> 4;            // 0: lanes 0-15, 1: lanes 16-31
  const int mn   = lane & 15;
  const int row0 = blockIdx.x * 16;
  const int col  = wave * 16 + mn;
  const bool colOK = col < Hc;
  const int  bcol  = colOK ? col : 0;    // clamp load column; store is guarded

  // lane's B stream: column bcol, K = kb + half*16 + e, contiguous in Wt
  const __bf16* brow = Wt + bcol * K + half * 16;
  const float2* arow = (const float2*)&As[mn][0];   // 8B-aligned pairs

  v8f acc = {};

  for (int kb = 0; kb < K; kb += 32) {
    __syncthreads();
    // async stage of the 16x32 fp32 A tile: 128 x b128, no VGPR round-trip
#pragma unroll
    for (int it = 0; it < (128 + NT - 1) / NT; ++it) {
      int g = tid + it * NT;
      if (g < 128) {
        int r  = g >> 3;                 // 8 float4 per 32-float row
        int c4 = (g & 7) * 4;
        unsigned long long ga =
            (unsigned long long)(uintptr_t)(A + (row0 + r) * lda + kb + c4);
        unsigned int la = (unsigned int)(uintptr_t)&As[r][c4];
        asm volatile("global_load_async_to_lds_b128 %0, %1, off"
                     :: "v"(la), "v"(ga) : "memory");
      }
    }
    asm volatile("s_wait_asynccnt 0" ::: "memory");
    __syncthreads();

    // A fragment: lane half 0 -> K {0..15}, half 1 -> K {16..31}
    v16bf a;
#pragma unroll
    for (int p = 0; p < 8; ++p) {
      int k = 2 * p + half * 16;
      float2 f = arow[k >> 1];           // ds_load_b64
      a[2 * p]     = (__bf16)f.x;
      a[2 * p + 1] = (__bf16)f.y;
    }
    // B fragment: one aligned 32-byte vector load
    v16bf b = *(const v16bf*)(brow + kb);

    acc = __builtin_amdgcn_wmma_f32_16x16x32_bf16(
        false, a, false, b, (short)0, acc, false, false);
  }

  if (colOK) {
#pragma unroll
    for (int r = 0; r < 8; ++r) {
      int row = row0 + r + half * 8;     // C/D layout: VGPR r -> M = r + 8*half
      float v = acc[r];
      if constexpr (ADD_BIAS) v += bias[col];
      C[row * ldc + col] = v;
    }
  }
}

// ---------------- edge scatter: agg[col] += h[row] * dinv[row]*dinv[col] ----------------
__global__ void scatter_kernel(const float* __restrict__ h,
                               const int* __restrict__ rowi,
                               const int* __restrict__ coli,
                               const float* __restrict__ dinv,
                               float* __restrict__ agg, int e)
{
  int gid  = blockIdx.x * blockDim.x + threadIdx.x;
  int edge = gid >> 5;
  int lane = gid & 31;
  if (edge >= e) return;
  int r = __builtin_amdgcn_readfirstlane(rowi[edge]);
  int c = __builtin_amdgcn_readfirstlane(coli[edge]);
  float w = dinv[r] * dinv[c];
  float4 v = ((const float4*)(h + r * HDIM))[lane];
  float* dst = agg + c * HDIM + lane * 4;
  unsafeAtomicAdd(dst + 0, v.x * w);
  unsafeAtomicAdd(dst + 1, v.y * w);
  unsafeAtomicAdd(dst + 2, v.z * w);
  unsafeAtomicAdd(dst + 3, v.w * w);
}

// ---------------- finish: self-loop term + bias, ReLU, write into JK buffer ----------------
__global__ void finish_kernel(const float* __restrict__ agg,
                              const float* __restrict__ h,
                              const float* __restrict__ dinv,
                              const float* __restrict__ bias,
                              float* __restrict__ xs, int n, int off)
{
  int idx = blockIdx.x * blockDim.x + threadIdx.x;
  if (idx >= n * HDIM) return;
  int i = idx >> 7, j = idx & 127;
  float di = dinv[i];
  float v = agg[idx] + h[idx] * di * di + bias[j];
  xs[i * XSW + off + j] = fmaxf(v, 0.0f);
}

// ---------------- launch ----------------

static inline size_t alignup(size_t v) { return (v + 255) & ~(size_t)255; }

extern "C" void kernel_launch(void* const* d_in, const int* in_sizes, int n_in,
                              void* d_out, int out_size, void* d_ws, size_t ws_size,
                              hipStream_t stream)
{
  const float* x    = (const float*)d_in[0];
  const int*   ei   = (const int*)d_in[1];
  const float* W0   = (const float*)d_in[2];
  const float* b0   = (const float*)d_in[3];
  const float* W1   = (const float*)d_in[4];
  const float* b1   = (const float*)d_in[5];
  const float* W2   = (const float*)d_in[6];
  const float* b2   = (const float*)d_in[7];
  const float* Wout = (const float*)d_in[8];
  const float* bout = (const float*)d_in[9];

  const int D = 256;
  const int N = in_sizes[0] / D;    // 50000 (multiple of 16)
  const int E = in_sizes[1] / 2;    // 800000

  const int* rowi = ei;             // edge_index[0]
  const int* coli = ei + E;         // edge_index[1]

  // workspace layout: dinv | h | agg | xs (fp32) | Wt0..WtO (bf16), 256B aligned
  char* ws = (char*)d_ws;
  float* dinv = (float*)ws;                 ws += alignup((size_t)N * sizeof(float));
  float* h    = (float*)ws;                 ws += alignup((size_t)N * HDIM * sizeof(float));
  float* agg  = (float*)ws;                 ws += alignup((size_t)N * HDIM * sizeof(float));
  float* xs   = (float*)ws;                 ws += alignup((size_t)N * XSW * sizeof(float));
  __bf16* Wt0 = (__bf16*)ws;                ws += alignup((size_t)HDIM * D * sizeof(__bf16));
  __bf16* Wt1 = (__bf16*)ws;                ws += alignup((size_t)HDIM * HDIM * sizeof(__bf16));
  __bf16* Wt2 = (__bf16*)ws;                ws += alignup((size_t)HDIM * HDIM * sizeof(__bf16));
  __bf16* WtO = (__bf16*)ws;

  // prep: transpose+convert weights to bf16
  wconv_kernel<<<(D * HDIM + 255) / 256, 256, 0, stream>>>(W0, Wt0, D, HDIM);
  wconv_kernel<<<(HDIM * HDIM + 255) / 256, 256, 0, stream>>>(W1, Wt1, HDIM, HDIM);
  wconv_kernel<<<(HDIM * HDIM + 255) / 256, 256, 0, stream>>>(W2, Wt2, HDIM, HDIM);
  wconv_kernel<<<(XSW * 40 + 255) / 256, 256, 0, stream>>>(Wout, WtO, XSW, 40);

  // gcn_norm: deg = 1 + in-degree; dinv = rsqrt(deg)
  deg_init_kernel <<<(N + 255) / 256, 256, 0, stream>>>(dinv, N);
  deg_count_kernel<<<(E + 255) / 256, 256, 0, stream>>>(coli, dinv, E);
  deg_rsqrt_kernel<<<(N + 255) / 256, 256, 0, stream>>>(dinv, N);

  const __bf16* Wts[3] = {Wt0, Wt1, Wt2};
  const float*  bs[3]  = {b0, b1, b2};

  for (int l = 0; l < 3; ++l) {
    const float* Ain; int lda, K;
    if (l == 0) { Ain = x;                   lda = D;   K = D;    }
    else        { Ain = xs + (l - 1) * HDIM; lda = XSW; K = HDIM; }

    // h = X @ W_l   (8 col tiles of 16 -> 8 waves / 256 threads)
    wmma_gemm_kernel<8, false><<<N / 16, 8 * 32, 0, stream>>>(
        Ain, lda, Wts[l], nullptr, h, HDIM, K, HDIM);

    hipMemsetAsync(agg, 0, (size_t)N * HDIM * sizeof(float), stream);

    scatter_kernel<<<(E * 32 + 255) / 256, 256, 0, stream>>>(
        h, rowi, coli, dinv, agg, E);

    finish_kernel<<<(N * HDIM + 255) / 256, 256, 0, stream>>>(
        agg, h, dinv, bs[l], xs, N, l * HDIM);
  }

  // out = xs @ Wout + bout   (40 cols -> 3 col tiles / 96 threads, guarded)
  wmma_gemm_kernel<3, true><<<N / 16, 3 * 32, 0, stream>>>(
      xs, XSW, WtO, bout, (float*)d_out, 40, XSW, 40);
}
